// LinearAttention_34445637714088
// MI455X (gfx1250) — compile-verified, matrix operand order checked by reading, not run on hardware
//
#include <hip/hip_runtime.h>

// ---------------- problem constants ----------------
#define B_   8
#define T_   8192
#define S_   8192
#define F_   64
#define D_   64
#define CH_  32                 // S-chunks per batch in phase 1
#define RPC_ (S_/CH_)           // 256 rows per chunk
#define NBLK_ (RPC_/32)         // 8 s-blocks of 32 rows per chunk
#define EPSF 1e-9f
#define PHI_SCALE 0.0625f       // 1/sqrt(2*128)

typedef __attribute__((ext_vector_type(16))) _Float16 v16h;
typedef __attribute__((ext_vector_type(8)))  _Float16 v8h;
typedef __attribute__((ext_vector_type(16))) __bf16   v16bf;
typedef __attribute__((ext_vector_type(8)))  __bf16   v8bf;
typedef __attribute__((ext_vector_type(8)))  float    v8f;
typedef __attribute__((ext_vector_type(4)))  float    v4f;
typedef __attribute__((ext_vector_type(4)))  unsigned int u32x4;
typedef __attribute__((ext_vector_type(8)))  int      i32x8;
typedef __attribute__((ext_vector_type(4)))  int      i32x4;

__device__ __forceinline__ v16h joinh(v8h lo, v8h hi) {
    return __builtin_shufflevector(lo, hi, 0,1,2,3,4,5,6,7,8,9,10,11,12,13,14,15);
}
__device__ __forceinline__ v16bf joinbf(v8bf lo, v8bf hi) {
    return __builtin_shufflevector(lo, hi, 0,1,2,3,4,5,6,7,8,9,10,11,12,13,14,15);
}

// Tensor Data Mover: flat copy of 2048 f32 (8 KB) global -> LDS.
// D# per CDNA5 ISA:
//   group0: [1:0]=count=1, [63:32]=lds_addr, [120:64]=global_addr, [127:126]=type=2
//   group1: data_size=4B (code 2), tensor_dim0=tile_dim0=2048, dim0_stride=2048,
//           tile_dim1/2=0 (unused), no padding, no multicast, no iteration
// This toolchain exposes the 6-arg builtin:
//   (u32x4 g0, i32x8 g1, i32x4 g2, i32x4 g3, i32x8 pad, i32 cpol)
__device__ __forceinline__ void tdm_load_8kb(unsigned lds_off, const void* gptr) {
    unsigned long long ga = (unsigned long long)(uintptr_t)gptr;
    u32x4 g0 = {};
    g0[0] = 1u;                                              // count=1 (valid)
    g0[1] = lds_off;                                         // LDS byte address
    g0[2] = (unsigned)(ga & 0xFFFFFFFFull);                  // global_addr[31:0]
    g0[3] = (unsigned)((ga >> 32) & 0x01FFFFFFull)           // global_addr[56:32]
            | (2u << 30);                                    // type=2 ("image")
    i32x8 g1 = {};
    g1[0] = (int)(2u << 16);                                 // data_size = 4 bytes
    g1[1] = (int)(2048u << 16);                              // tensor_dim0 = 2048
    g1[2] = 0;                                               // tensor_dim1 = 0
    g1[3] = (int)(2048u << 16);                              // tile_dim0 = 2048
    g1[4] = 0;                                               // tile_dim1/2 = 0
    g1[5] = 2048;                                            // tensor_dim0_stride
    g1[6] = 0;
    g1[7] = 0;
    i32x4 z4 = {};
    i32x8 z8v = {};
    __builtin_amdgcn_tensor_load_to_lds(g0, g1, z4, z4, z8v, 0);
}

// ---------------- kernel 0: omega f32 -> f16 (row-major 128x64) ----------------
__global__ __launch_bounds__(256) void k_prep(const float* __restrict__ w,
                                              _Float16* __restrict__ o) {
    int i = blockIdx.x * 256 + threadIdx.x;   // 8192 elements total
    o[i] = (_Float16)w[i];
}

// ---------------- kernel 1: partial kv = phi(K)^T @ [V,1] per (b, chunk) -----
// grid = B_*CH_, block = 256 (8 waves). Wave w owns feature tiles m=w (pos) and
// m=w+8 (neg). Key/value tiles stream in via double-buffered TDM async DMA.
__global__ __launch_bounds__(256) void k_phase1(const float* __restrict__ keyg,
                                                const float* __restrict__ valg,
                                                const _Float16* __restrict__ omg,
                                                float* __restrict__ kvpart) {
    __shared__ alignas(16) float    stage[2][2][2048]; // [buf][key|val] f32 8KB tiles
    __shared__ alignas(16) _Float16 k_lds[32 * 72];    // key tile, f16, pitch 72
    __shared__ alignas(16) __bf16   v1T[80 * 40];      // [v,1]^T, bf16, pitch 40
    __shared__ alignas(32) float    sq_part[256];
    __shared__ alignas(32) float    sq[32];            // |x|^2/2 per key row

    const int tid  = threadIdx.x;
    const int wave = tid >> 5;
    const int lane = tid & 31;
    const int l16  = lane & 15;
    const int lh   = lane >> 4;

    const int b = blockIdx.x >> 5;        // / CH_
    const int c = blockIdx.x & (CH_ - 1);
    const int sBase0 = c * RPC_;

    // LDS byte offsets of the staging buffers (generic-pointer low 32 bits)
    const unsigned stK0 = (unsigned)(uintptr_t)&stage[0][0][0];
    const unsigned stV0 = (unsigned)(uintptr_t)&stage[0][1][0];
    const unsigned stK1 = (unsigned)(uintptr_t)&stage[1][0][0];
    const unsigned stV1 = (unsigned)(uintptr_t)&stage[1][1][0];

    // constant part of v1T: ones row (d=64), zero pad rows 65..79
    for (int i = 64 * 40 + tid; i < 80 * 40; i += 256) {
        int d = i / 40, s = i % 40;
        v1T[i] = (d == 64 && s < 32) ? (__bf16)1.0f : (__bf16)0.0f;
    }

    // omega B-tiles for this wave's feature tile (m = wave*16 .. +15), K=f
    v16h omB[2];
#pragma unroll
    for (int kk = 0; kk < 2; ++kk) {
        const _Float16* rp = omg + (wave * 16 + l16) * F_;
        omB[kk] = joinh(*(const v8h*)(rp + kk * 32 + lh * 8),
                        *(const v8h*)(rp + kk * 32 + 16 + lh * 8));
    }

    v8f accP[5], accN[5];
    v8f z8 = {};
#pragma unroll
    for (int nt = 0; nt < 5; ++nt) { accP[nt] = z8; accN[nt] = z8; }

    // prologue: async-DMA block 0 into buffer 0
    if (wave == 0) {
        tdm_load_8kb(stK0, keyg + ((long)b * S_ + sBase0) * F_);
        tdm_load_8kb(stV0, valg + ((long)b * S_ + sBase0) * D_);
    }

    for (int blk = 0; blk < NBLK_; ++blk) {
        const int buf = blk & 1;
        if (wave == 0) __builtin_amdgcn_s_wait_tensorcnt(0);  // current tiles landed
        __syncthreads();   // publish stage[buf]; prev iteration's LDS reads done

        // kick off the DMA for the next block while we compute on this one
        if (wave == 0 && blk + 1 < NBLK_) {
            const int sN = sBase0 + (blk + 1) * 32;
            tdm_load_8kb(buf ? stK0 : stK1, keyg + ((long)b * S_ + sN) * F_);
            tdm_load_8kb(buf ? stV0 : stV1, valg + ((long)b * S_ + sN) * D_);
        }

        // ---- key tile: LDS f32 stage -> f16 k_lds + |x|^2 partials ----
        {
            int r = tid >> 3, f0 = (tid & 7) * 8;
            const float* sp = &stage[buf][0][r * 64 + f0];
            v4f a = *(const v4f*)sp;
            v4f bb = *(const v4f*)(sp + 4);
            float fv[8] = {a.x, a.y, a.z, a.w, bb.x, bb.y, bb.z, bb.w};
            v8h h = {};
            float ss = 0.f;
#pragma unroll
            for (int j = 0; j < 8; ++j) { ss += fv[j] * fv[j]; h[j] = (_Float16)fv[j]; }
            *(v8h*)&k_lds[r * 72 + f0] = h;
            sq_part[tid] = ss;
        }
        // ---- value tile: LDS f32 stage -> transposed bf16 v1T ----
#pragma unroll
        for (int i = 0; i < 8; ++i) {
            int idx = tid + i * 256;
            int d = idx & 63, s = idx >> 6;
            v1T[d * 40 + s] = (__bf16)stage[buf][1][s * 64 + d];
        }
        __syncthreads();
        if (tid < 32) {
            float a = 0.f;
#pragma unroll
            for (int j = 0; j < 8; ++j) a += sq_part[tid * 8 + j];
            sq[tid] = 0.5f * a;
        }
        __syncthreads();

        // ---- xw = key_tile(16x32f) x omega^T : two s-tiles, K=64 in 2 steps ----
        v8f xw0 = z8, xw1 = z8;
#pragma unroll
        for (int kk = 0; kk < 2; ++kk) {
            const _Float16* rp0 = &k_lds[(0 * 16 + l16) * 72];
            const _Float16* rp1 = &k_lds[(1 * 16 + l16) * 72];
            v16h a0 = joinh(*(const v8h*)(rp0 + kk * 32 + lh * 8),
                            *(const v8h*)(rp0 + kk * 32 + 16 + lh * 8));
            v16h a1 = joinh(*(const v8h*)(rp1 + kk * 32 + lh * 8),
                            *(const v8h*)(rp1 + kk * 32 + 16 + lh * 8));
            xw0 = __builtin_amdgcn_wmma_f32_16x16x32_f16(false, a0, false, omB[kk],
                                                         (short)0, xw0, false, false);
            xw1 = __builtin_amdgcn_wmma_f32_16x16x32_f16(false, a1, false, omB[kk],
                                                         (short)0, xw1, false, false);
        }

        // ---- phi: D-layout of xw (lane<->m, vgpr<->s) == A-layout of kv GEMM ----
        v16bf Apos = {}, Aneg = {};
        v8f sv0 = *(const v8f*)&sq[lh * 8];
        v8f sv1 = *(const v8f*)&sq[16 + lh * 8];
#pragma unroll
        for (int i = 0; i < 8; ++i) {
            float x0 = xw0[i], e0 = sv0[i];
            Apos[i]     = (__bf16)((__expf( x0 - e0) + EPSF) * PHI_SCALE);
            Aneg[i]     = (__bf16)((__expf(-x0 - e0) + EPSF) * PHI_SCALE);
            float x1 = xw1[i], e1 = sv1[i];
            Apos[8 + i] = (__bf16)((__expf( x1 - e1) + EPSF) * PHI_SCALE);
            Aneg[8 + i] = (__bf16)((__expf(-x1 - e1) + EPSF) * PHI_SCALE);
        }

        // ---- kv accumulate: phi^T (16m x 32s) x v1 (32s x 16d), 5 d-tiles ----
#pragma unroll
        for (int nt = 0; nt < 5; ++nt) {
            const __bf16* vp = &v1T[(nt * 16 + l16) * 40];
            v16bf Bv = joinbf(*(const v8bf*)(vp + lh * 8),
                              *(const v8bf*)(vp + 16 + lh * 8));
            accP[nt] = __builtin_amdgcn_wmma_f32_16x16x32_bf16(false, Apos, false, Bv,
                                                               (short)0, accP[nt], false, false);
            accN[nt] = __builtin_amdgcn_wmma_f32_16x16x32_bf16(false, Aneg, false, Bv,
                                                               (short)0, accN[nt], false, false);
        }
    }

    // ---- write f32 partial kv (256x65) ----
#pragma unroll
    for (int nt = 0; nt < 5; ++nt) {
        int d = nt * 16 + l16;
        if (d < 65) {
            long base0 = (long)(b * CH_ + c) * 256 * 65 + d;
#pragma unroll
            for (int r = 0; r < 8; ++r) {
                int mP = wave * 16 + lh * 8 + r;
                kvpart[base0 + (long)mP * 65]         = accP[nt][r];
                kvpart[base0 + (long)(128 + mP) * 65] = accN[nt][r];
            }
        }
    }
}

// ---------------- kernel 2: reduce partials -> kvT bf16 [b][d(80)][m(256)] ----
__global__ __launch_bounds__(256) void k_reduce(const float* __restrict__ kvpart,
                                                __bf16* __restrict__ kvT) {
    int bd = blockIdx.x;            // 0 .. B_*80-1
    int b = bd / 80, d = bd % 80;
    int m = threadIdx.x;
    float s = 0.f;
    if (d < 65) {
        for (int c = 0; c < CH_; ++c)
            s += kvpart[(long)(b * CH_ + c) * 256 * 65 + (long)m * 65 + d];
    }
    kvT[((long)b * 80 + d) * 256 + m] = (__bf16)s;
}

// ---------------- kernel 3: out = (phi(Q) @ kv)[:, :64] / (...)[:, 64] --------
// grid = B_ * T_/128, block = 256. Wave w owns t-tile w (16 rows of the 128-row
// block). xw^T = omega x q^T gives lane<->t, vgpr<->m -> directly the B-layout
// of qkv^T = kvT x phi^T. No LDS transpose of phi needed.
__global__ __launch_bounds__(256) void k_phase2(const float* __restrict__ qry,
                                                const _Float16* __restrict__ omg,
                                                const __bf16* __restrict__ kvT,
                                                float* __restrict__ outg) {
    __shared__ alignas(16) _Float16 q_lds[128 * 72];
    __shared__ alignas(32) float    sq_part[256];
    __shared__ alignas(32) float    sq[128];
    __shared__ alignas(16) float    o_lds[128 * 65];

    const int tid  = threadIdx.x;
    const int wave = tid >> 5;
    const int lane = tid & 31;
    const int l16  = lane & 15;
    const int lh   = lane >> 4;

    const int b  = blockIdx.x >> 6;          // / (T_/128)
    const int t0 = (blockIdx.x & 63) * 128;

    // ---- load query block (128x64), f32->f16 + row square sums ----
    {
        int r = tid >> 1, f0 = (tid & 1) * 32;
        const float* gp = qry + ((long)b * T_ + t0 + r) * F_ + f0;
        float ss = 0.f;
#pragma unroll
        for (int j = 0; j < 4; ++j) {
            v4f a = *(const v4f*)(gp + j * 8);
            v4f bb = *(const v4f*)(gp + j * 8 + 4);
            float fv[8] = {a.x, a.y, a.z, a.w, bb.x, bb.y, bb.z, bb.w};
            v8h h = {};
#pragma unroll
            for (int k = 0; k < 8; ++k) { ss += fv[k] * fv[k]; h[k] = (_Float16)fv[k]; }
            *(v8h*)&q_lds[r * 72 + f0 + j * 8] = h;
        }
        sq_part[tid] = ss;
    }
    __syncthreads();
    if (tid < 128) sq[tid] = 0.5f * (sq_part[2 * tid] + sq_part[2 * tid + 1]);
    __syncthreads();

    // ---- xw^T = omega(16m x 32f) x q^T(32f x 16t), 8 m-tiles, K=64 ----
    v8f z8 = {};
    v16h Bq[2];
#pragma unroll
    for (int kk = 0; kk < 2; ++kk) {
        const _Float16* qr = &q_lds[(wave * 16 + l16) * 72];
        Bq[kk] = joinh(*(const v8h*)(qr + kk * 32 + lh * 8),
                       *(const v8h*)(qr + kk * 32 + 16 + lh * 8));
    }
    v8f xw[8];
#pragma unroll
    for (int mt = 0; mt < 8; ++mt) {
        v8f cacc = z8;
#pragma unroll
        for (int kk = 0; kk < 2; ++kk) {
            const _Float16* orp = omg + (mt * 16 + l16) * F_;
            v16h Ao = joinh(*(const v8h*)(orp + kk * 32 + lh * 8),
                            *(const v8h*)(orp + kk * 32 + 16 + lh * 8));
            cacc = __builtin_amdgcn_wmma_f32_16x16x32_f16(false, Ao, false, Bq[kk],
                                                          (short)0, cacc, false, false);
        }
        xw[mt] = cacc;
    }

    // ---- phi -> bf16 B-tiles (K=m). posB[ks] covers m=32ks..+31; neg shifted 128 ----
    float sv = sq[wave * 16 + l16];
    v16bf posB[4], negB[4];
#pragma unroll
    for (int ks = 0; ks < 4; ++ks) {
        v16bf up = {}, un = {};
#pragma unroll
        for (int i = 0; i < 8; ++i) {
            float x0 = xw[2 * ks][i];
            up[i]     = (__bf16)((__expf( x0 - sv) + EPSF) * PHI_SCALE);
            un[i]     = (__bf16)((__expf(-x0 - sv) + EPSF) * PHI_SCALE);
            float x1 = xw[2 * ks + 1][i];
            up[8 + i] = (__bf16)((__expf( x1 - sv) + EPSF) * PHI_SCALE);
            un[8 + i] = (__bf16)((__expf(-x1 - sv) + EPSF) * PHI_SCALE);
        }
        posB[ks] = up; negB[ks] = un;
    }

    // ---- qkv^T = kvT(80d x 256m) x phi^T(256m x 16t): 5 d-tiles, 8 K-steps ----
    v8f qacc[5];
#pragma unroll
    for (int dt = 0; dt < 5; ++dt) qacc[dt] = z8;
#pragma unroll
    for (int dt = 0; dt < 5; ++dt) {
        const __bf16* ar = kvT + ((long)b * 80 + dt * 16 + l16) * 256;
#pragma unroll
        for (int ks = 0; ks < 8; ++ks) {
            v16bf Ak = joinbf(*(const v8bf*)(ar + ks * 32 + lh * 8),
                              *(const v8bf*)(ar + ks * 32 + 16 + lh * 8));
            v16bf Bp = (ks < 4) ? posB[ks] : negB[ks - 4];
            qacc[dt] = __builtin_amdgcn_wmma_f32_16x16x32_bf16(false, Ak, false, Bp,
                                                               (short)0, qacc[dt], false, false);
        }
    }

    // ---- normalizer (d=64 lives in dt=4, r=0, lh=0), broadcast across lh ----
    float nv = qacc[4][0];
    float nrm = __builtin_bit_cast(float,
        __builtin_amdgcn_ds_bpermute(l16 << 2, __builtin_bit_cast(int, nv)));
    float rn = 1.0f / nrm;

    // ---- normalize, transpose via LDS, coalesced store ----
#pragma unroll
    for (int dt = 0; dt < 4; ++dt)
#pragma unroll
        for (int r = 0; r < 8; ++r) {
            int d = dt * 16 + lh * 8 + r;
            o_lds[(wave * 16 + l16) * 65 + d] = qacc[dt][r] * rn;
        }
    __syncthreads();
#pragma unroll
    for (int i = 0; i < 32; ++i) {
        int idx = tid + i * 256;
        int t = idx >> 6, d = idx & 63;
        outg[((long)b * T_ + t0 + t) * 64 + d] = o_lds[t * 65 + d];
    }
}

// ---------------- host launch ----------------
extern "C" void kernel_launch(void* const* d_in, const int* in_sizes, int n_in,
                              void* d_out, int out_size, void* d_ws, size_t ws_size,
                              hipStream_t stream) {
    const float* query = (const float*)d_in[0];
    const float* value = (const float*)d_in[1];
    const float* key   = (const float*)d_in[2];
    const float* omega = (const float*)d_in[3];
    float* out = (float*)d_out;

    char* ws = (char*)d_ws;
    _Float16* omg16 = (_Float16*)ws;                       // 128*64*2      = 16 KB
    __bf16*   kvT   = (__bf16*)(ws + 16384);               // 8*80*256*2    = 320 KB
    float*    kvpart = (float*)(ws + 16384 + 327680);      // 32*8*256*65*4 = 17 MB

    k_prep  <<<32,              256, 0, stream>>>(omega, omg16);
    k_phase1<<<B_ * CH_,        256, 0, stream>>>(key, value, omg16, kvpart);
    k_reduce<<<B_ * 80,         256, 0, stream>>>(kvpart, kvT);
    k_phase2<<<B_ * (T_ / 128), 256, 0, stream>>>(query, omg16, kvT, out);
}
